// _PerceiverResamplerModule_56358560858342
// MI455X (gfx1250) — matrix operand order repaired and checked by reading, not compiled, not run on hardware
//
#include <hip/hip_runtime.h>
#include <math.h>

typedef unsigned short u16;
typedef __attribute__((ext_vector_type(16))) __bf16 v16bf;
typedef __attribute__((ext_vector_type(8)))  __bf16 v8bf;
typedef __attribute__((ext_vector_type(8)))  float  v8f;

union AU { v16bf v; v8bf h[2]; };

__device__ __forceinline__ float bf2f(u16 u) {
    unsigned int x = ((unsigned int)u) << 16;
    float f; __builtin_memcpy(&f, &x, 4); return f;
}
__device__ __forceinline__ u16 f2bf(float f) {
    unsigned int x; __builtin_memcpy(&x, &f, 4);
    unsigned int r = x + 0x7FFFu + ((x >> 16) & 1u);
    return (u16)(r >> 16);
}

// ---------------------------------------------------------------------------
// init latents: lat[bt*64+i][d] = latents[i][d]   (2048 x 1024 f32)
// ---------------------------------------------------------------------------
__global__ void init_lat_k(const float* __restrict__ latents, float* __restrict__ lat) {
    size_t idx = (size_t)blockIdx.x * 256 + threadIdx.x;   // 2,097,152 total
    int d = (int)(idx & 1023);
    int r = (int)(idx >> 10);
    lat[idx] = latents[(size_t)(r & 63) * 1024 + d];
}

// ---------------------------------------------------------------------------
// f32 [K,N] -> bf16 [N,K] transpose-convert (LDS tiled). K,N multiples of 32.
// ---------------------------------------------------------------------------
__global__ __launch_bounds__(256) void transpose_f32_bf16_k(
        const float* __restrict__ src, u16* __restrict__ dst, int K, int N) {
    __shared__ float tile[32][33];
    int kb = blockIdx.x * 32, nb = blockIdx.y * 32;
    int tx = threadIdx.x, ty = threadIdx.y;      // 32 x 8
    for (int r = 0; r < 32; r += 8)
        tile[ty + r][tx] = src[(size_t)(kb + ty + r) * N + (nb + tx)];
    __syncthreads();
    for (int r = 0; r < 32; r += 8)
        dst[(size_t)(nb + ty + r) * K + (kb + tx)] = f2bf(tile[tx][ty + r]);
}

// ---------------------------------------------------------------------------
// LayerNorm over last dim (1024). One block (256 thr) per row.
// dstA: contiguous bf16 (or null). dstB: bf16 at mapped row
//   row2 = (r/per)*gstride + goff + (r%per)   (or null). dstF: f32 (or null).
// ---------------------------------------------------------------------------
__global__ __launch_bounds__(256) void ln_k(
        const float* __restrict__ src, const float* __restrict__ g,
        const float* __restrict__ b, u16* __restrict__ dstA,
        u16* __restrict__ dstB, int per, int gstride, int goff,
        float* __restrict__ dstF) {
    __shared__ float red[256];
    int r = blockIdx.x, tid = threadIdx.x;
    const float* x = src + (size_t)r * 1024;
    float v[4]; float s = 0.f;
    for (int j = 0; j < 4; ++j) { v[j] = x[tid + 256 * j]; s += v[j]; }
    red[tid] = s; __syncthreads();
    for (int o = 128; o > 0; o >>= 1) { if (tid < o) red[tid] += red[tid + o]; __syncthreads(); }
    float mean = red[0] * (1.0f / 1024.0f);
    __syncthreads();
    float sq = 0.f;
    for (int j = 0; j < 4; ++j) { float d = v[j] - mean; sq += d * d; }
    red[tid] = sq; __syncthreads();
    for (int o = 128; o > 0; o >>= 1) { if (tid < o) red[tid] += red[tid + o]; __syncthreads(); }
    float rstd = rsqrtf(red[0] * (1.0f / 1024.0f) + 1e-5f);
    size_t rowA = (size_t)r * 1024;
    size_t rowB = ((size_t)(r / per) * gstride + goff + (r % per)) * 1024;
    for (int j = 0; j < 4; ++j) {
        int d = tid + 256 * j;
        float y = (v[j] - mean) * rstd * g[d] + b[d];
        if (dstA) dstA[rowA + d] = f2bf(y);
        if (dstB) dstB[rowB + d] = f2bf(y);
        if (dstF) dstF[rowA + d] = y;
    }
}

// ---------------------------------------------------------------------------
// C[M,N] = A[M,K](bf16,row-major) x Bt[N,K](bf16, B transposed), f32 accum via
// v_wmma_f32_16x16x32_bf16. Block = 256 thr = 8 waves (2x4); each wave holds a
// 4x2 grid of 16x16 accumulator tiles (64x32 per wave -> 128x128 per block):
// 6 operand loads feed 8 WMMAs per K-step (0.75 loads per WMMA).
// MODE 0: store f32   1: store bf16   2: Cf += acc (residual)   3: gelu->bf16
// ---------------------------------------------------------------------------
template<int MODE>
__global__ __launch_bounds__(256) void gemm_bf16_k(
        const u16* __restrict__ A, const u16* __restrict__ Bt,
        float* __restrict__ Cf, u16* __restrict__ Cu, int M, int N, int K) {
    (void)M;
    int w = threadIdx.x >> 5, lane = threadIdx.x & 31;
    int half = lane >> 4, ln = lane & 15;
    int mbase = blockIdx.y * 128 + (w & 1) * 64;   // 2 wave-rows x 64
    int nbase = blockIdx.x * 128 + (w >> 1) * 32;  // 4 wave-cols x 32
    const u16* ap[4]; const u16* bp[2];
    #pragma unroll
    for (int i = 0; i < 4; ++i) ap[i] = A + (size_t)(mbase + i * 16 + ln) * K;
    #pragma unroll
    for (int j = 0; j < 2; ++j) bp[j] = Bt + (size_t)(nbase + j * 16 + ln) * K + half * 16;
    v8f acc[4][2];
    #pragma unroll
    for (int i = 0; i < 4; ++i)
        #pragma unroll
        for (int j = 0; j < 2; ++j) acc[i][j] = (v8f){};
    for (int k0 = 0; k0 < K; k0 += 32) {
        AU a[4]; v16bf b[2];
        #pragma unroll
        for (int i = 0; i < 4; ++i) {
            a[i].h[0] = *(const v8bf*)(ap[i] + k0 + half * 16);      // K: h*16..+7
            a[i].h[1] = *(const v8bf*)(ap[i] + k0 + half * 16 + 8);  // K: h*16+8..+15
        }
        #pragma unroll
        for (int j = 0; j < 2; ++j) b[j] = *(const v16bf*)(bp[j] + k0);  // K: h*16..+15
        #pragma unroll
        for (int i = 0; i < 4; ++i)
            #pragma unroll
            for (int j = 0; j < 2; ++j)
                acc[i][j] = __builtin_amdgcn_wmma_f32_16x16x32_bf16(
                    false, a[i].v, false, b[j], (short)0, acc[i][j], false, false);
    }
    #pragma unroll
    for (int i = 0; i < 4; ++i) {
        #pragma unroll
        for (int j = 0; j < 2; ++j) {
            int orow = mbase + i * 16 + half * 8;
            int ocol = nbase + j * 16 + ln;
            #pragma unroll
            for (int r = 0; r < 8; ++r) {
                size_t idx = (size_t)(orow + r) * N + ocol;
                float v = acc[i][j][r];
                if (MODE == 0)      Cf[idx] = v;
                else if (MODE == 1) Cu[idx] = f2bf(v);
                else if (MODE == 2) Cf[idx] += v;
                else {
                    float ge = 0.5f * v * (1.0f + erff(v * 0.70710678118654752f));
                    Cu[idx] = f2bf(ge);
                }
            }
        }
    }
}

// ---------------------------------------------------------------------------
// Attention: one block per (bt, head). q f32 [2048,1024] (scale applied here),
// kv bf16 [32,1088,2048] (k cols 0..1023, v cols 1024..2047).
// Each wave owns latent rows i = w, w+8, ... ; softmax over 1088 keys.
// Output o bf16 [2048,1024].
// ---------------------------------------------------------------------------
__global__ __launch_bounds__(256) void attn_k(
        const float* __restrict__ q, const u16* __restrict__ kv,
        u16* __restrict__ o) {
    __shared__ float qs[64 * 64];
    __shared__ float srow[8 * 1088];
    int bt = blockIdx.x >> 4, head = blockIdx.x & 15;
    int tid = threadIdx.x;
    for (int t = tid; t < 4096; t += 256) {
        int i = t >> 6, d = t & 63;
        qs[t] = q[(size_t)(bt * 64 + i) * 1024 + head * 64 + d] * 0.125f;
    }
    __syncthreads();
    int w = tid >> 5, lane = tid & 31;
    const u16* kbase = kv + (size_t)bt * 1088 * 2048 + head * 64;
    const u16* vbase = kbase + 1024;
    float* sr = srow + w * 1088;
    for (int i = w; i < 64; i += 8) {
        const float* qi = qs + i * 64;
        float m = -1e30f;
        for (int j = lane; j < 1088; j += 32) {
            const u16* kr = kbase + (size_t)j * 2048;
            float s = 0.f;
            #pragma unroll
            for (int d = 0; d < 64; ++d) s += qi[d] * bf2f(kr[d]);
            sr[j] = s; m = fmaxf(m, s);
        }
        for (int off = 16; off; off >>= 1) m = fmaxf(m, __shfl_xor(m, off, 32));
        float l = 0.f;
        for (int j = lane; j < 1088; j += 32) {
            float p = __expf(sr[j] - m); sr[j] = p; l += p;
        }
        for (int off = 16; off; off >>= 1) l += __shfl_xor(l, off, 32);
        float inv = 1.0f / l;
        for (int c = 0; c < 4; ++c) {
            float accv[16];
            #pragma unroll
            for (int dd = 0; dd < 16; ++dd) accv[dd] = 0.f;
            for (int j = lane; j < 1088; j += 32) {
                float p = sr[j];
                const u16* vr = vbase + (size_t)j * 2048 + c * 16;
                #pragma unroll
                for (int dd = 0; dd < 16; ++dd) accv[dd] += p * bf2f(vr[dd]);
            }
            #pragma unroll
            for (int dd = 0; dd < 16; ++dd)
                for (int off = 16; off; off >>= 1)
                    accv[dd] += __shfl_xor(accv[dd], off, 32);
            if (lane == 0) {
                size_t ob = (size_t)(bt * 64 + i) * 1024 + head * 64 + c * 16;
                for (int dd = 0; dd < 16; ++dd) o[ob + dd] = f2bf(accv[dd] * inv);
            }
        }
    }
}

// ---------------------------------------------------------------------------
extern "C" void kernel_launch(void* const* d_in, const int* in_sizes, int n_in,
                              void* d_out, int out_size, void* d_ws, size_t ws_size,
                              hipStream_t stream) {
    (void)in_sizes; (void)n_in; (void)out_size; (void)ws_size;
    const float* x       = (const float*)d_in[0];   // [32768,1024]
    const float* latents = (const float*)d_in[1];   // [64,1024]
    const float* nm_g    = (const float*)d_in[2];
    const float* nm_b    = (const float*)d_in[3];
    const float* nl_g    = (const float*)d_in[4];
    const float* nl_b    = (const float*)d_in[5];
    const float* wq      = (const float*)d_in[6];   // [6,1024,1024]
    const float* wkv     = (const float*)d_in[7];   // [6,1024,2048]
    const float* wo      = (const float*)d_in[8];   // [6,1024,1024]
    const float* ff_g    = (const float*)d_in[9];
    const float* ff_b    = (const float*)d_in[10];
    const float* w1      = (const float*)d_in[11];  // [6,1024,4096]
    const float* w2      = (const float*)d_in[12];  // [6,4096,1024]
    const float* fin_g   = (const float*)d_in[13];
    const float* fin_b   = (const float*)d_in[14];

    char* ws = (char*)d_ws;
    size_t off = 0;
    float* lat   = (float*)(ws + off); off += (size_t)2048 * 1024 * 4;   // 8 MiB
    float* qbuf  = (float*)(ws + off); off += (size_t)2048 * 1024 * 4;   // 8 MiB
    u16* lnlat   = (u16*)(ws + off);   off += (size_t)2048 * 1024 * 2;   // 4 MiB
    u16* ffin    = (u16*)(ws + off);   off += (size_t)2048 * 1024 * 2;   // 4 MiB
    u16* obf     = (u16*)(ws + off);   off += (size_t)2048 * 1024 * 2;   // 4 MiB
    u16* h1      = (u16*)(ws + off);   off += (size_t)2048 * 4096 * 2;   // 16 MiB
    u16* xnl     = (u16*)(ws + off);   off += (size_t)32 * 1088 * 1024 * 2; // 68 MiB
    u16* kvb     = (u16*)(ws + off);   off += (size_t)32 * 1088 * 2048 * 2; // 136 MiB
    u16* wq_t    = (u16*)(ws + off);   off += (size_t)1024 * 1024 * 2;
    u16* wo_t    = (u16*)(ws + off);   off += (size_t)1024 * 1024 * 2;
    u16* wkv_t   = (u16*)(ws + off);   off += (size_t)2048 * 1024 * 2;
    u16* w1_t    = (u16*)(ws + off);   off += (size_t)4096 * 1024 * 2;
    u16* w2_t    = (u16*)(ws + off);   off += (size_t)1024 * 4096 * 2;

    dim3 tb(32, 8);
    init_lat_k<<<8192, 256, 0, stream>>>(latents, lat);

    for (int i = 0; i < 6; ++i) {
        // per-layer weight conversion: f32 [K,N] -> bf16 [N,K]
        transpose_f32_bf16_k<<<dim3(32, 32),  tb, 0, stream>>>(wq  + (size_t)i * 1024 * 1024, wq_t, 1024, 1024);
        transpose_f32_bf16_k<<<dim3(32, 64),  tb, 0, stream>>>(wkv + (size_t)i * 1024 * 2048, wkv_t, 1024, 2048);
        transpose_f32_bf16_k<<<dim3(32, 32),  tb, 0, stream>>>(wo  + (size_t)i * 1024 * 1024, wo_t, 1024, 1024);
        transpose_f32_bf16_k<<<dim3(32, 128), tb, 0, stream>>>(w1  + (size_t)i * 1024 * 4096, w1_t, 1024, 4096);
        transpose_f32_bf16_k<<<dim3(128, 32), tb, 0, stream>>>(w2  + (size_t)i * 4096 * 1024, w2_t, 4096, 1024);

        // LN(x) -> xnl media rows [bt*1088 + m]
        ln_k<<<32768, 256, 0, stream>>>(x, nm_g + i * 1024, nm_b + i * 1024,
                                        nullptr, xnl, 1024, 1088, 0, nullptr);
        // LN(lat) -> lnlat (contig) and xnl latent rows [bt*1088 + 1024 + j]
        ln_k<<<2048, 256, 0, stream>>>(lat, nl_g + i * 1024, nl_b + i * 1024,
                                       lnlat, xnl, 64, 1088, 1024, nullptr);
        // q = lnlat @ wq   (f32 out)  M=2048 N=1024 K=1024
        gemm_bf16_k<0><<<dim3(8, 16), 256, 0, stream>>>(lnlat, wq_t, qbuf, nullptr, 2048, 1024, 1024);
        // kv = xnl @ wkv   (bf16 out)  M=34816 N=2048 K=1024
        gemm_bf16_k<1><<<dim3(16, 272), 256, 0, stream>>>(xnl, wkv_t, nullptr, kvb, 34816, 2048, 1024);
        // attention -> obf (bf16)
        attn_k<<<512, 256, 0, stream>>>(qbuf, kvb, obf);
        // lat += obf @ wo   M=2048 N=1024 K=1024
        gemm_bf16_k<2><<<dim3(8, 16), 256, 0, stream>>>(obf, wo_t, lat, nullptr, 2048, 1024, 1024);
        // FF
        ln_k<<<2048, 256, 0, stream>>>(lat, ff_g + i * 1024, ff_b + i * 1024,
                                       ffin, nullptr, 1, 1, 0, nullptr);
        // h1 = gelu(ffin @ w1)  M=2048 N=4096 K=1024
        gemm_bf16_k<3><<<dim3(32, 16), 256, 0, stream>>>(ffin, w1_t, nullptr, h1, 2048, 4096, 1024);
        // lat += h1 @ w2   M=2048 N=1024 K=4096
        gemm_bf16_k<2><<<dim3(8, 16), 256, 0, stream>>>(h1, w2_t, lat, nullptr, 2048, 1024, 4096);
    }
    // final LN -> f32 output
    ln_k<<<2048, 256, 0, stream>>>(lat, fin_g, fin_b, nullptr, nullptr, 1, 1, 0, (float*)d_out);
}
